// FASAAttention_32839319945666
// MI455X (gfx1250) — matrix operand order repaired and checked by reading, not compiled, not run on hardware
//
#include <hip/hip_runtime.h>
#include <math.h>
#include <stdint.h>

// ---------------- problem constants ----------------
#define S_LEN   2048
#define HID     512
#define NHEAD   8
#define DHEAD   64
#define NSEL    256
#define SCALE   0.125f      // 1/sqrt(64)

// ---------------- WMMA types ----------------
typedef __bf16 v16bf __attribute__((ext_vector_type(16)));
typedef float  v8f   __attribute__((ext_vector_type(8)));
typedef int    v4i   __attribute__((ext_vector_type(4)));
typedef unsigned short ushort8 __attribute__((ext_vector_type(8)));

union FragU { ushort8 h[2]; v16bf v; };

__device__ __forceinline__ unsigned short f2bf(float f) {
    unsigned u = __float_as_uint(f);
    u += 0x7FFFu + ((u >> 16) & 1u);          // round-to-nearest-even
    return (unsigned short)(u >> 16);
}
__device__ __forceinline__ float bf2f(unsigned short h) {
    return __uint_as_float(((unsigned)h) << 16);
}
__device__ __forceinline__ unsigned f2ord(float f) {
    unsigned u = __float_as_uint(f);
    return (u & 0x80000000u) ? ~u : (u | 0x80000000u);
}

// ---------------- async global -> LDS staging (CDNA5 path) ----------------
#if defined(__has_builtin)
# if __has_builtin(__builtin_amdgcn_global_load_async_to_lds_b128)
#  define HAVE_ASYNC_LDS 1
# endif
# if __has_builtin(__builtin_amdgcn_s_wait_asynccnt)
#  define HAVE_WAIT_ASYNC 1
# endif
#endif
#ifndef HAVE_ASYNC_LDS
# define HAVE_ASYNC_LDS 0
#endif
#ifndef HAVE_WAIT_ASYNC
# define HAVE_WAIT_ASYNC 0
#endif

__device__ __forceinline__ void cp16_g2l(const unsigned short* g, unsigned short* l) {
#if HAVE_ASYNC_LDS
    __builtin_amdgcn_global_load_async_to_lds_b128(
        reinterpret_cast<__attribute__((address_space(1))) v4i*>(reinterpret_cast<uintptr_t>(g)),
        reinterpret_cast<__attribute__((address_space(3))) v4i*>((unsigned)reinterpret_cast<uintptr_t>(l)),
        0, 0);
#else
    *(ushort8*)l = *(const ushort8*)g;        // global_load_b128 + ds_store_b128
#endif
}
__device__ __forceinline__ void cp_wait() {
#if HAVE_ASYNC_LDS
# if HAVE_WAIT_ASYNC
    __builtin_amdgcn_s_wait_asynccnt(0);
# else
    asm volatile("s_wait_asynccnt 0x0" ::: "memory");
# endif
#endif
}

// ---------------- fragment loaders (ISA §7.12.2 layouts) ----------------
// A fragment: 16x32 bf16 tile, row-major source, row stride ld (halves).
__device__ __forceinline__ v16bf load_a_frag(const unsigned short* base, int ld, int lane) {
    int m = lane & 15, sel = (lane >> 4) & 1;
    const unsigned short* row = base + m * ld + sel * 8;
    FragU f;
    f.h[0] = *(const ushort8*)(row);
    f.h[1] = *(const ushort8*)(row + 16);
    return f.v;
}
// B fragment: 32x16 bf16, B[k][n] = W[n][k] with W row-major stride ld.
__device__ __forceinline__ v16bf load_b_frag(const unsigned short* base, int ld, int lane) {
    int n = lane & 15, sel = (lane >> 4) & 1;
    const unsigned short* row = base + n * ld + sel * 8;
    FragU f;
    f.h[0] = *(const ushort8*)(row);
    f.h[1] = *(const ushort8*)(row + 16);
    return f.v;
}

__device__ __forceinline__ v8f vzero8() {
    v8f z = {0.f,0.f,0.f,0.f,0.f,0.f,0.f,0.f};
    return z;
}

// ---------------- kernel 0: fp32 -> bf16 ----------------
__global__ void to_bf16_kernel(const float* __restrict__ src, unsigned short* __restrict__ dst, int n) {
    int i = blockIdx.x * blockDim.x + threadIdx.x;
    if (i < n) dst[i] = f2bf(src[i]);
}

// ---------------- LDS-staged block GEMM: 64(M) x 128(N), K=512 ----------------
#define BM 64
#define BN 128
#define BK 32
#define NKSTEP (HID / BK)   // 16

// 8 waves: wm = wid>>1 in [0,4), wn = wid&1 in [0,2); wave tile = 16 x 64.
__device__ __forceinline__ void block_gemm_64x128(
    const unsigned short* __restrict__ Ag,   // [BM][HID] row-major (block's M rows)
    const unsigned short* __restrict__ Bg,   // [BN][HID] row-major (block's N rows of W)
    unsigned short (*Asm)[BM * BK],
    unsigned short (*Bsm)[BN * BK],
    v8f acc[4])
{
    int tid  = threadIdx.x;
    int wid  = tid >> 5, lane = tid & 31;
    int wm   = wid >> 1, wn = wid & 1;

    auto stage = [&](int ks, int buf) {
        int k0 = ks * BK;
        {   // A tile: 64x32 halves = 4 KB -> 256 x 16B chunks, 1 per thread
            int ar = tid >> 2, ac = (tid & 3) * 8;
            cp16_g2l(Ag + ar * HID + k0 + ac, Asm[buf] + ar * BK + ac);
        }
#pragma unroll
        for (int q = 0; q < 2; ++q) {  // B tile: 128x32 halves = 8 KB -> 2 chunks/thread
            int c  = tid + q * 256;
            int br = c >> 2, bc = (c & 3) * 8;
            cp16_g2l(Bg + br * HID + k0 + bc, Bsm[buf] + br * BK + bc);
        }
    };

    stage(0, 0);
    cp_wait();
    __syncthreads();
    for (int ks = 0; ks < NKSTEP; ++ks) {
        int cur = ks & 1;
        if (ks + 1 < NKSTEP) stage(ks + 1, cur ^ 1);
        const unsigned short* As = Asm[cur];
        const unsigned short* Bs = Bsm[cur];
        v16bf a = load_a_frag(As + wm * 16 * BK, BK, lane);
#pragma unroll
        for (int j = 0; j < 4; ++j) {
            v16bf b = load_b_frag(Bs + (wn * 64 + j * 16) * BK, BK, lane);
            acc[j] = __builtin_amdgcn_wmma_f32_16x16x32_bf16(false, a, false, b,
                                                             (short)0, acc[j], false, false);
        }
        cp_wait();
        __syncthreads();
    }
}

// ---------------- kernel 1: Q/K/V projections ----------------
// Y[s,o] = sum_c x[s,c] * W[o,c]; outputs head-major [H][S][D] bf16.
__global__ void __launch_bounds__(256) qkv_gemm_kernel(
        const unsigned short* __restrict__ xb,
        const unsigned short* __restrict__ wq,
        const unsigned short* __restrict__ wk,
        const unsigned short* __restrict__ wv,
        unsigned short* __restrict__ Qb,
        unsigned short* __restrict__ Kb,
        unsigned short* __restrict__ Vb) {
    __shared__ __align__(16) unsigned short Asm[2][BM * BK];
    __shared__ __align__(16) unsigned short Bsm[2][BN * BK];

    int bid = blockIdx.x;                 // 3 * 32 * 4 = 384 blocks
    int mat = bid >> 7;
    int r   = bid & 127;
    int mt  = r >> 2, nt = r & 3;
    const unsigned short* W = (mat == 0) ? wq : (mat == 1) ? wk : wv;
    unsigned short*       Y = (mat == 0) ? Qb : (mat == 1) ? Kb : Vb;

    v8f acc[4] = { vzero8(), vzero8(), vzero8(), vzero8() };
    block_gemm_64x128(xb + mt * BM * HID, W + nt * BN * HID, Asm, Bsm, acc);

    int wid = threadIdx.x >> 5, lane = threadIdx.x & 31;
    int wm = wid >> 1, wn = wid & 1;
    int n = lane & 15, mb = ((lane >> 4) & 1) * 8;
#pragma unroll
    for (int j = 0; j < 4; ++j) {
        int o = nt * BN + wn * 64 + j * 16 + n;
        int h = o >> 6, d = o & 63;
#pragma unroll
        for (int i = 0; i < 8; ++i) {
            int s = mt * BM + wm * 16 + mb + i;
            Y[(h * S_LEN + s) * DHEAD + d] = f2bf(acc[j][i]);
        }
    }
}

// ---------------- kernel 2: importance + exact top-256 ----------------
__global__ void importance_topk_kernel(const unsigned short* __restrict__ Qb,
                                       const unsigned short* __restrict__ Kb,
                                       int* __restrict__ topidx) {
    extern __shared__ char smem[];
    float*    imp  = (float*)smem;                       // [16][2048]  128 KB
    unsigned* hist = (unsigned*)(smem + 16 * S_LEN * 4); // [16][256]   16 KB

    int wid = threadIdx.x >> 5, lane = threadIdx.x & 31;
    int h  = blockIdx.x >> 7;
    int qt = blockIdx.x & 127;
    int q0 = qt * 16;

    // importance GEMM: A = Q[h] rows q0..q0+15, dims 0..31 (single K=32 step)
    const unsigned short* Abase = Qb + (h * S_LEN + q0) * DHEAD;
    v16bf a = load_a_frag(Abase, DHEAD, lane);
    int n = lane & 15, mbase = ((lane >> 4) & 1) * 8;
    for (int kt = wid; kt < 128; kt += 16) {
        const unsigned short* Bbase = Kb + (h * S_LEN + kt * 16) * DHEAD;
        v16bf b = load_b_frag(Bbase, DHEAD, lane);
        v8f acc = vzero8();
        acc = __builtin_amdgcn_wmma_f32_16x16x32_bf16(false, a, false, b,
                                                      (short)0, acc, false, false);
#pragma unroll
        for (int i = 0; i < 8; i++)
            imp[(mbase + i) * S_LEN + kt * 16 + n] = acc[i];
    }
    __syncthreads();

    // radix top-256 for row r = wid
    float*    row = imp  + wid * S_LEN;
    unsigned* hst = hist + wid * 256;
    unsigned prefix = 0;
    int need = NSEL;
    for (int pos = 3; pos >= 0; --pos) {
        for (int i = lane; i < 256; i += 32) hst[i] = 0u;
        __syncthreads();
        int shift = pos * 8;
        for (int i = lane; i < S_LEN; i += 32) {
            unsigned u = f2ord(row[i]);
            bool cand = (pos == 3) || ((u >> (shift + 8)) == prefix);
            if (cand) atomicAdd(&hst[(u >> shift) & 0xFF], 1u);
        }
        __syncthreads();
        unsigned cum = 0; int chosen = 0;
        for (int b = 255; b >= 0; --b) {
            unsigned c = hst[b];
            if (cum + c >= (unsigned)need) { chosen = b; break; }
            cum += c;
        }
        need  -= (int)cum;
        prefix = (prefix << 8) | (unsigned)chosen;
        __syncthreads();
    }

    unsigned T = prefix;
    unsigned lmlt = (1u << lane) - 1u;
    int cnt = 0, eq_need = need;
    int* outp = topidx + (h * S_LEN + q0 + wid) * NSEL;
    for (int base = 0; base < S_LEN; base += 32) {
        unsigned u = f2ord(row[base + lane]);
        bool gt = u > T;
        bool eq = (u == T);
        unsigned emask = (unsigned)__ballot(eq);
        int epos = __popc(emask & lmlt);
        bool take = gt || (eq && (epos < eq_need));
        unsigned tmask = (unsigned)__ballot(take);
        int pos = cnt + __popc(tmask & lmlt);
        if (take && pos < NSEL) outp[pos] = base + lane;
        cnt += __popc(tmask);
        int et = __popc(emask);
        eq_need -= (et < eq_need) ? et : eq_need;
    }
}

// ---------------- kernel 3: gathered attention ----------------
#define ATT_NW 8
__global__ void attn_kernel(const unsigned short* __restrict__ Qb,
                            const unsigned short* __restrict__ Kb,
                            const unsigned short* __restrict__ Vb,
                            const int* __restrict__ topidx,
                            unsigned short* __restrict__ attnout) {
    __shared__ float qv[ATT_NW][DHEAD];
    __shared__ float pj[ATT_NW][NSEL];
    __shared__ int   ij[ATT_NW][NSEL];
    int wid = threadIdx.x >> 5, lane = threadIdx.x & 31;
    int p = blockIdx.x * ATT_NW + wid;
    int h = p >> 11, q = p & (S_LEN - 1);

    const unsigned short* qr = Qb + (h * S_LEN + q) * DHEAD;
    {
        unsigned u = *(const unsigned*)(qr + lane * 2);
        qv[wid][lane * 2]     = bf2f((unsigned short)(u & 0xFFFFu));
        qv[wid][lane * 2 + 1] = bf2f((unsigned short)(u >> 16));
    }
    __syncthreads();

    const int* ti = topidx + (h * S_LEN + q) * NSEL;
    float mx = -3.0e38f;
    float sc[8]; int idxr[8];
#pragma unroll
    for (int t = 0; t < 8; t++) {
        int j = t * 32 + lane;
        int idx = ti[j];
        idxr[t] = idx;
        const unsigned short* kr = Kb + (h * S_LEN + idx) * DHEAD;
        __builtin_prefetch(Vb + (h * S_LEN + idx) * DHEAD, 0, 0);  // warm V rows
        float s = 0.f;
        for (int d0 = 0; d0 < DHEAD; d0 += 8) {
            ushort8 kk = *(const ushort8*)(kr + d0);
#pragma unroll
            for (int e = 0; e < 8; e++) s += qv[wid][d0 + e] * bf2f(kk[e]);
        }
        s *= SCALE;
        sc[t] = s;
        mx = fmaxf(mx, s);
    }
#pragma unroll
    for (int o = 16; o >= 1; o >>= 1) mx = fmaxf(mx, __shfl_xor(mx, o, 32));
    float sum = 0.f;
#pragma unroll
    for (int t = 0; t < 8; t++) {
        float e = __expf(sc[t] - mx);
        sum += e;
        pj[wid][t * 32 + lane] = e;
        ij[wid][t * 32 + lane] = idxr[t];
    }
#pragma unroll
    for (int o = 16; o >= 1; o >>= 1) sum += __shfl_xor(sum, o, 32);
    float rinv = 1.0f / sum;
    __syncthreads();

    int d0 = lane * 2;
    float a0 = 0.f, a1 = 0.f;
    for (int j = 0; j < NSEL; j++) {
        int idx = ij[wid][j];
        float pv = pj[wid][j];
        const unsigned short* vr = Vb + (h * S_LEN + idx) * DHEAD;
        unsigned u = *(const unsigned*)(vr + d0);
        a0 += pv * bf2f((unsigned short)(u & 0xFFFFu));
        a1 += pv * bf2f((unsigned short)(u >> 16));
    }
    a0 *= rinv; a1 *= rinv;
    unsigned short* orow = attnout + q * HID + h * DHEAD;
    orow[d0]     = f2bf(a0);
    orow[d0 + 1] = f2bf(a1);
}

// ---------------- kernel 4: output projection (fp32 result) ----------------
__global__ void __launch_bounds__(256) out_gemm_kernel(
        const unsigned short* __restrict__ ab,
        const unsigned short* __restrict__ wo,
        float* __restrict__ out) {
    __shared__ __align__(16) unsigned short Asm[2][BM * BK];
    __shared__ __align__(16) unsigned short Bsm[2][BN * BK];

    int bid = blockIdx.x;                 // 32 * 4 = 128 blocks
    int mt  = bid >> 2, nt = bid & 3;

    v8f acc[4] = { vzero8(), vzero8(), vzero8(), vzero8() };
    block_gemm_64x128(ab + mt * BM * HID, wo + nt * BN * HID, Asm, Bsm, acc);

    int wid = threadIdx.x >> 5, lane = threadIdx.x & 31;
    int wm = wid >> 1, wn = wid & 1;
    int n = lane & 15, mb = ((lane >> 4) & 1) * 8;
#pragma unroll
    for (int j = 0; j < 4; ++j) {
        int o = nt * BN + wn * 64 + j * 16 + n;
#pragma unroll
        for (int i = 0; i < 8; ++i) {
            int s = mt * BM + wm * 16 + mb + i;
            out[s * HID + o] = acc[j][i];
        }
    }
}

// ---------------- host launcher ----------------
extern "C" void kernel_launch(void* const* d_in, const int* in_sizes, int n_in,
                              void* d_out, int out_size, void* d_ws, size_t ws_size,
                              hipStream_t stream) {
    const float* x  = (const float*)d_in[0];
    const float* wq = (const float*)d_in[1];
    const float* wk = (const float*)d_in[2];
    const float* wv = (const float*)d_in[3];
    const float* wo = (const float*)d_in[4];

    char* ws = (char*)d_ws;
    size_t off = 0;
    auto alloc = [&](size_t bytes) { char* p = ws + off; off += (bytes + 255) & ~size_t(255); return p; };

    unsigned short* xb  = (unsigned short*)alloc((size_t)S_LEN * HID * 2);
    unsigned short* wqb = (unsigned short*)alloc((size_t)HID * HID * 2);
    unsigned short* wkb = (unsigned short*)alloc((size_t)HID * HID * 2);
    unsigned short* wvb = (unsigned short*)alloc((size_t)HID * HID * 2);
    unsigned short* wob = (unsigned short*)alloc((size_t)HID * HID * 2);
    unsigned short* Qb  = (unsigned short*)alloc((size_t)NHEAD * S_LEN * DHEAD * 2);
    unsigned short* Kb  = (unsigned short*)alloc((size_t)NHEAD * S_LEN * DHEAD * 2);
    unsigned short* Vb  = (unsigned short*)alloc((size_t)NHEAD * S_LEN * DHEAD * 2);
    int*            ti  = (int*)alloc((size_t)NHEAD * S_LEN * NSEL * 4);
    unsigned short* ao  = (unsigned short*)alloc((size_t)S_LEN * HID * 2);

    int nx = S_LEN * HID, nw = HID * HID;
    to_bf16_kernel<<<(nx + 255) / 256, 256, 0, stream>>>(x,  xb,  nx);
    to_bf16_kernel<<<(nw + 255) / 256, 256, 0, stream>>>(wq, wqb, nw);
    to_bf16_kernel<<<(nw + 255) / 256, 256, 0, stream>>>(wk, wkb, nw);
    to_bf16_kernel<<<(nw + 255) / 256, 256, 0, stream>>>(wv, wvb, nw);
    to_bf16_kernel<<<(nw + 255) / 256, 256, 0, stream>>>(wo, wob, nw);

    // 1) Q/K/V projections: 3 mats x (2048/64) x (512/128) blocks
    qkv_gemm_kernel<<<3 * 32 * 4, 256, 0, stream>>>(xb, wqb, wkb, wvb, Qb, Kb, Vb);

    // 2) importance + top-256 (128 KB imp + 16 KB hist dynamic LDS)
    size_t lds = (size_t)16 * S_LEN * 4 + (size_t)16 * 256 * 4;
    (void)hipFuncSetAttribute((const void*)importance_topk_kernel,
                              hipFuncAttributeMaxDynamicSharedMemorySize, (int)lds);
    importance_topk_kernel<<<NHEAD * (S_LEN / 16), 512, lds, stream>>>(Qb, Kb, ti);

    // 3) gathered attention: 1 wave per (h,q)
    attn_kernel<<<(NHEAD * S_LEN) / ATT_NW, ATT_NW * 32, 0, stream>>>(Qb, Kb, Vb, ti, ao);

    // 4) output projection -> fp32 d_out
    out_gemm_kernel<<<32 * 4, 256, 0, stream>>>(ao, wob, (float*)d_out);
}
